// seg_network_9998683865706
// MI455X (gfx1250) — matrix operand order repaired and checked by reading, not compiled, not run on hardware
//
#include <hip/hip_runtime.h>
#include <hip/hip_bf16.h>

// ---------------------------------------------------------------------------
// PointNet++ segmentation forward (N=131072, K_SEG=50) for gfx1250 (MI455X).
// Heavy fp3 MLP layers use v_wmma_f32_16x16x32_bf16 (bf16 in, fp32 accum),
// register-blocked: A held in VGPRs across all K-steps, 2 column tiles in
// flight per wave (independent accumulator chains).
// Irregular stages (FPS, ball query, kNN, BN stats) are wave32 VALU kernels.
// ---------------------------------------------------------------------------

#define NPTS 131072
#define NC1  512
#define NC2  128

typedef __attribute__((ext_vector_type(16))) __bf16 v16bf;
typedef __attribute__((ext_vector_type(8)))  float  v8f;

// ---------------------------------------------------------------- utilities
__global__ void zero_f32(float* __restrict__ p, int n) {
  int i = blockIdx.x * blockDim.x + threadIdx.x;
  if (i < n) p[i] = 0.0f;
}

// ------------------------------------------------------------ farthest-point
// Single workgroup, sequential FPS (matches jnp.argmax first-index tie-break).
__global__ void fps_kernel(const float* __restrict__ pts, int n, int k,
                           int* __restrict__ idx, float* __restrict__ cent,
                           float* __restrict__ dists) {
  __shared__ float sd[1024];
  __shared__ int   si[1024];
  __shared__ int   s_last;
  const int tid = threadIdx.x;
  for (int i = tid; i < n; i += blockDim.x) dists[i] = 3.4e38f;
  if (tid == 0) {
    idx[0] = 0; s_last = 0;
    cent[0] = pts[0]; cent[1] = pts[1]; cent[2] = pts[2];
  }
  __syncthreads();
  for (int j = 1; j < k; ++j) {
    const int last = s_last;
    const float lx = pts[last * 3], ly = pts[last * 3 + 1], lz = pts[last * 3 + 2];
    float bd = -1.0f; int bi = 0;
    for (int i = tid; i < n; i += blockDim.x) {
      float dx = pts[i * 3] - lx, dy = pts[i * 3 + 1] - ly, dz = pts[i * 3 + 2] - lz;
      float d = dx * dx + dy * dy + dz * dz;
      float nd = fminf(dists[i], d);
      dists[i] = nd;
      if (nd > bd) { bd = nd; bi = i; }
    }
    sd[tid] = bd; si[tid] = bi;
    __syncthreads();
    for (int s = blockDim.x >> 1; s > 0; s >>= 1) {
      if (tid < s) {
        if (sd[tid + s] > sd[tid] ||
            (sd[tid + s] == sd[tid] && si[tid + s] < si[tid])) {
          sd[tid] = sd[tid + s]; si[tid] = si[tid + s];
        }
      }
      __syncthreads();
    }
    if (tid == 0) {
      int b = si[0];
      idx[j] = b; s_last = b;
      cent[j * 3] = pts[b * 3]; cent[j * 3 + 1] = pts[b * 3 + 1];
      cent[j * 3 + 2] = pts[b * 3 + 2];
    }
    __syncthreads();
  }
}

// -------------------------------------------------------------- ball query
// One wave32 per centroid; collects first gk in-radius points in index order
// via ballot prefix-sum. Pads with -1.
__global__ void ball_query(const float* __restrict__ pts, int n,
                           const float* __restrict__ cent, float r2, int gk,
                           int* __restrict__ grp) {
  const int c = blockIdx.x;
  const int lane = threadIdx.x;
  const float cx = cent[c * 3], cy = cent[c * 3 + 1], cz = cent[c * 3 + 2];
  int cnt = 0;
  for (int base = 0; base < n && cnt < gk; base += 32) {
    int i = base + lane;
    bool in = false;
    if (i < n) {
      float dx = pts[i * 3] - cx, dy = pts[i * 3 + 1] - cy, dz = pts[i * 3 + 2] - cz;
      in = (dx * dx + dy * dy + dz * dz) <= r2;
    }
    unsigned mask = (unsigned)__ballot(in);
    int pre = __popc(mask & ((1u << lane) - 1u));
    if (in) {
      int slot = cnt + pre;
      if (slot < gk) grp[c * gk + slot] = i;
    }
    cnt += __popc(mask);
  }
  if (cnt > gk) cnt = gk;
  for (int j = cnt + lane; j < gk; j += 32) grp[c * gk + j] = -1;
}

// ------------------------------------------------- grouped MLP + max pooling
// One block per centroid group; per-point 3-layer MLP, ReLU, LDS max-pool.
// ReLU outputs are >= 0 so int atomicMax on float bits is a valid max.
template <int CIN, int H1, int H2, int COUT>
__global__ void sa_mlp(const float* __restrict__ pts, const float* __restrict__ feats,
                       const float* __restrict__ cent, const int* __restrict__ grp,
                       int gk,
                       const float* __restrict__ w1, const float* __restrict__ b1,
                       const float* __restrict__ w2, const float* __restrict__ b2,
                       const float* __restrict__ w3, const float* __restrict__ b3,
                       float* __restrict__ fout) {
  constexpr int FDIM = CIN - 3;
  __shared__ int smax[COUT];
  const int c = blockIdx.x;
  for (int j = threadIdx.x; j < COUT; j += blockDim.x) smax[j] = 0;
  __syncthreads();
  const float cx = cent[c * 3], cy = cent[c * 3 + 1], cz = cent[c * 3 + 2];
  for (int p = threadIdx.x; p < gk; p += blockDim.x) {
    const int pi = grp[c * gk + p];
    if (pi >= 0) {
      float h1[H1];
      const float x0 = pts[pi * 3] - cx, x1 = pts[pi * 3 + 1] - cy,
                  x2 = pts[pi * 3 + 2] - cz;
      #pragma unroll 4
      for (int j = 0; j < H1; ++j)
        h1[j] = b1[j] + x0 * w1[j] + x1 * w1[H1 + j] + x2 * w1[2 * H1 + j];
      if constexpr (FDIM > 0) {
        #pragma unroll 1
        for (int i = 0; i < FDIM; ++i) {
          const float xi = feats[(long)pi * FDIM + i];
          #pragma unroll 4
          for (int j = 0; j < H1; ++j) h1[j] += xi * w1[(3 + i) * H1 + j];
        }
      }
      for (int j = 0; j < H1; ++j) h1[j] = fmaxf(h1[j], 0.0f);
      float h2[H2];
      for (int j = 0; j < H2; ++j) h2[j] = b2[j];
      #pragma unroll 1
      for (int i = 0; i < H1; ++i) {
        const float v = h1[i];
        #pragma unroll 4
        for (int j = 0; j < H2; ++j) h2[j] += v * w2[i * H2 + j];
      }
      for (int j = 0; j < H2; ++j) h2[j] = fmaxf(h2[j], 0.0f);
      #pragma unroll 1
      for (int j = 0; j < COUT; ++j) {
        float a = b3[j];
        #pragma unroll 4
        for (int i = 0; i < H2; ++i) a += h2[i] * w3[i * COUT + j];
        a = fmaxf(a, 0.0f);
        atomicMax(&smax[j], __float_as_int(a));
      }
    }
  }
  __syncthreads();
  for (int j = threadIdx.x; j < COUT; j += blockDim.x)
    fout[(long)c * COUT + j] = __int_as_float(smax[j]);
}

// --------------------------------------------------------- small dense GEMM
__global__ void dense_f32(const float* __restrict__ Xin, int K,
                          const float* __restrict__ W, const float* __restrict__ b,
                          float* __restrict__ out, int M, int Nout, int relu) {
  const int e = blockIdx.x * blockDim.x + threadIdx.x;
  if (e >= M * Nout) return;
  const int r = e / Nout, c = e % Nout;
  float acc = b[c];
  const float* x = Xin + (long)r * K;
  for (int k = 0; k < K; ++k) acc += x[k] * W[(long)k * Nout + c];
  if (relu) acc = fmaxf(acc, 0.0f);
  out[e] = acc;
}

__global__ void rowmax(const float* __restrict__ in, int M, int C,
                       float* __restrict__ out) {
  const int c = blockIdx.x * blockDim.x + threadIdx.x;
  if (c >= C) return;
  float m = -3.4e38f;
  for (int r = 0; r < M; ++r) m = fmaxf(m, in[(long)r * C + c]);
  out[c] = m;
}

// ------------------------------------------------------------------- 3-NN
__global__ void knn3(const float* __restrict__ tgt, int nt,
                     const float* __restrict__ src, int ns,
                     int* __restrict__ idx3, float* __restrict__ w3) {
  const int t = blockIdx.x * blockDim.x + threadIdx.x;
  if (t >= nt) return;
  const float tx = tgt[t * 3], ty = tgt[t * 3 + 1], tz = tgt[t * 3 + 2];
  float bd0 = 1e30f, bd1 = 1e30f, bd2 = 1e30f;
  int bi0 = 0, bi1 = 0, bi2 = 0;
  for (int s = 0; s < ns; ++s) {
    float dx = src[s * 3] - tx, dy = src[s * 3 + 1] - ty, dz = src[s * 3 + 2] - tz;
    float d = dx * dx + dy * dy + dz * dz;
    if (d < bd0)      { bd2 = bd1; bi2 = bi1; bd1 = bd0; bi1 = bi0; bd0 = d; bi0 = s; }
    else if (d < bd1) { bd2 = bd1; bi2 = bi1; bd1 = d; bi1 = s; }
    else if (d < bd2) { bd2 = d; bi2 = s; }
  }
  float d0 = fmaxf(sqrtf(fmaxf(bd0, 1e-12f)), 1e-6f);
  float d1 = fmaxf(sqrtf(fmaxf(bd1, 1e-12f)), 1e-6f);
  float d2 = fmaxf(sqrtf(fmaxf(bd2, 1e-12f)), 1e-6f);
  float w0 = 1.0f / (d0 * d0), w1 = 1.0f / (d1 * d1), w2 = 1.0f / (d2 * d2);
  float s = w0 + w1 + w2;
  idx3[t * 3] = bi0; idx3[t * 3 + 1] = bi1; idx3[t * 3 + 2] = bi2;
  w3[t * 3] = w0 / s; w3[t * 3 + 1] = w1 / s; w3[t * 3 + 2] = w2 / s;
}

// ------------------------------------------- concat [tgtc | tgtf | interp]
// idx3==nullptr => broadcast single src row (fp1 case, k=min(1,3)=1, w==1).
__global__ void fp_concat(const float* __restrict__ tgtc,
                          const float* __restrict__ tgtf, int F,
                          const float* __restrict__ srcf, int C,
                          const int* __restrict__ idx3, const float* __restrict__ w3,
                          float* __restrict__ out, int nt, int stride) {
  const long e = (long)blockIdx.x * blockDim.x + threadIdx.x;
  if (e >= (long)nt * stride) return;
  const int t = (int)(e / stride);
  const int c = (int)(e % stride);
  float v;
  if (c < 3) v = tgtc[t * 3 + c];
  else if (c < 3 + F) v = tgtf[(long)t * F + (c - 3)];
  else {
    const int cc = c - 3 - F;
    if (idx3) {
      const int* id = idx3 + t * 3;
      const float* w = w3 + t * 3;
      v = w[0] * srcf[(long)id[0] * C + cc] +
          w[1] * srcf[(long)id[1] * C + cc] +
          w[2] * srcf[(long)id[2] * C + cc];
    } else {
      v = srcf[cc];
    }
  }
  out[e] = v;
}

// -------------------------------- fp3 input: [coords(3)|interp(128)|pad(29)]
__global__ void fp3_concat_bf16(const float* __restrict__ coords,
                                const float* __restrict__ f5,
                                const int* __restrict__ idx3,
                                const float* __restrict__ w3,
                                __bf16* __restrict__ X) {
  const long e = (long)blockIdx.x * blockDim.x + threadIdx.x;
  if (e >= (long)NPTS * 160) return;
  const long t = e / 160;
  const int c = (int)(e % 160);
  float v = 0.0f;
  if (c < 3) v = coords[t * 3 + c];
  else if (c < 131) {
    const int cc = c - 3;
    const int* id = idx3 + t * 3;
    const float* w = w3 + t * 3;
    v = w[0] * f5[(long)id[0] * 128 + cc] +
        w[1] * f5[(long)id[1] * 128 + cc] +
        w[2] * f5[(long)id[2] * 128 + cc];
  }
  X[e] = (__bf16)v;
}

// ---------------------------------------------------- BatchNorm statistics
// One thread per channel, 256-row chunk per block, atomicAdd partials.
__global__ void bn_stats(const float* __restrict__ Y, long M, int C,
                         float* __restrict__ acc) {
  const int c = threadIdx.x;
  if (c >= C) return;
  long r0 = (long)blockIdx.x * 256;
  long r1 = r0 + 256; if (r1 > M) r1 = M;
  float s = 0.0f, ss = 0.0f;
  for (long r = r0; r < r1; ++r) {
    const float v = Y[r * C + c];
    s += v; ss += v * v;
  }
  atomicAdd(&acc[c], s);
  atomicAdd(&acc[C + c], ss);
}

template <bool RELU, bool BF16OUT>
__global__ void bn_apply(const float* __restrict__ Y, const float* __restrict__ acc,
                         const float* __restrict__ g, const float* __restrict__ bt,
                         long M, int C, float* __restrict__ outF,
                         __bf16* __restrict__ outB) {
  const long e = (long)blockIdx.x * blockDim.x + threadIdx.x;
  if (e >= M * C) return;
  const int c = (int)(e % C);
  const float mean = acc[c] / (float)M;
  const float var = acc[C + c] / (float)M - mean * mean;
  float v = g[c] * (Y[e] - mean) * rsqrtf(var + 1e-5f) + bt[c];
  if (RELU) v = fmaxf(v, 0.0f);
  if (BF16OUT) outB[e] = (__bf16)v;
  else outF[e] = v;
}

// ------------------------------------ pack weights into WMMA B-lane layout
// B tile is 32x16 (KxN): lane l holds col N=l%16, K = 32*s + 8*(l/16) + (e&7) + 16*(e>>3).
__global__ void pack_B(const float* __restrict__ W, int Kreal, int Nreal,
                       int Kp, int Np, __bf16* __restrict__ out) {
  const long e = (long)blockIdx.x * blockDim.x + threadIdx.x;
  if (e >= (long)Kp * Np) return;
  const int eidx = (int)(e & 15);
  long tmp = e >> 4;
  const int lane = (int)(tmp & 31); tmp >>= 5;
  const int ksteps = Kp >> 5;
  const int s = (int)(tmp % ksteps);
  const int ct = (int)(tmp / ksteps);
  const int K = s * 32 + (lane < 16 ? 0 : 8) + (eidx & 7) + (eidx >> 3) * 16;
  const int Ncol = ct * 16 + (lane & 15);
  const float v = (K < Kreal && Ncol < Nreal) ? W[(long)K * Nreal + Ncol] : 0.0f;
  out[e] = (__bf16)v;
}

// ------------------------------------------------ WMMA GEMM: Y = X@W + b
// bf16 A/B, fp32 accumulate. One 16-row tile per wave; A for all K-steps is
// preloaded into VGPRs (KSTEPS is a template arg so the array stays in regs),
// then the wave sweeps column tiles two at a time (two independent WMMA
// accumulator chains). Cuts A traffic 8x vs one-tile-per-wave and doubles
// WMMA issue density. Packed B (<=40KB) stays resident in WGP$/L2.
template <int KSTEPS>
__global__ void wmma_dense(const __bf16* __restrict__ X,
                           const __bf16* __restrict__ Bp,
                           const float* __restrict__ bias,
                           float* __restrict__ Y, int M, int Np, int Nstore) {
  const int lane = threadIdx.x & 31;
  const int wave = threadIdx.x >> 5;
  const long rowTile = (long)blockIdx.x * (blockDim.x >> 5) + wave;
  if (rowTile >= (M >> 4)) return;  // wave-uniform: EXEC all-ones at wmma
  const int m = lane & 15;
  const int half = lane >> 4;
  const int colTiles = Np >> 4;
  constexpr int Kp = KSTEPS * 32;

  union AB { v16bf v; float4 q[2]; };

  // Preload the whole 16 x Kp A tile for this wave into registers.
  const __bf16* xrow = X + (rowTile * 16 + m) * (long)Kp + half * 8;
  AB a[KSTEPS];
  #pragma unroll
  for (int s = 0; s < KSTEPS; ++s) {
    a[s].q[0] = *(const float4*)(xrow + s * 32);
    a[s].q[1] = *(const float4*)(xrow + s * 32 + 16);
  }

  const long rbase = rowTile * 16 + half * 8;
  for (int ct = 0; ct < colTiles; ct += 2) {
    const __bf16* b0 = Bp + ((long)ct * KSTEPS * 32 + lane) * 16;
    const __bf16* b1 = b0 + (long)KSTEPS * 512;
    v8f acc0 = {};
    v8f acc1 = {};
    #pragma unroll
    for (int s = 0; s < KSTEPS; ++s) {
      AB bb0, bb1;
      bb0.q[0] = *(const float4*)(b0 + s * 512);
      bb0.q[1] = *(const float4*)(b0 + s * 512 + 8);
      bb1.q[0] = *(const float4*)(b1 + s * 512);
      bb1.q[1] = *(const float4*)(b1 + s * 512 + 8);
      acc0 = __builtin_amdgcn_wmma_f32_16x16x32_bf16(false, a[s].v, false, bb0.v,
                                                     (short)0, acc0, false, false);
      acc1 = __builtin_amdgcn_wmma_f32_16x16x32_bf16(false, a[s].v, false, bb1.v,
                                                     (short)0, acc1, false, false);
    }
    const int n0 = ct * 16 + m;
    const int n1 = n0 + 16;
    if (n0 < Nstore) {
      const float bv = bias[n0];
      #pragma unroll
      for (int r = 0; r < 8; ++r)
        Y[(rbase + r) * (long)Nstore + n0] = acc0[r] + bv;
    }
    if (n1 < Nstore) {
      const float bv = bias[n1];
      #pragma unroll
      for (int r = 0; r < 8; ++r)
        Y[(rbase + r) * (long)Nstore + n1] = acc1[r] + bv;
    }
  }
}

// ===========================================================================
extern "C" void kernel_launch(void* const* d_in, const int* in_sizes, int n_in,
                              void* d_out, int out_size, void* d_ws, size_t ws_size,
                              hipStream_t stream) {
  (void)in_sizes; (void)n_in; (void)out_size; (void)ws_size;
  // --- input pointers (flattened setup_inputs() order) ---
  const float* coords = (const float*)d_in[0];
  // d_in[1] = features (empty)
  const float* s1w0 = (const float*)d_in[2],  *s1b0 = (const float*)d_in[3];
  const float* s1w1 = (const float*)d_in[4],  *s1b1 = (const float*)d_in[5];
  const float* s1w2 = (const float*)d_in[6],  *s1b2 = (const float*)d_in[7];
  const float* s2w0 = (const float*)d_in[8],  *s2b0 = (const float*)d_in[9];
  const float* s2w1 = (const float*)d_in[10], *s2b1 = (const float*)d_in[11];
  const float* s2w2 = (const float*)d_in[12], *s2b2 = (const float*)d_in[13];
  const float* gw0 = (const float*)d_in[14], *gb0 = (const float*)d_in[15];
  const float* gw1 = (const float*)d_in[16], *gb1 = (const float*)d_in[17];
  const float* gw2 = (const float*)d_in[18], *gb2 = (const float*)d_in[19];
  const float* p1w0 = (const float*)d_in[20], *p1b0 = (const float*)d_in[21];
  const float* p1g0 = (const float*)d_in[22], *p1t0 = (const float*)d_in[23];
  const float* p1w1 = (const float*)d_in[24], *p1b1 = (const float*)d_in[25];
  const float* p1g1 = (const float*)d_in[26], *p1t1 = (const float*)d_in[27];
  const float* p2w0 = (const float*)d_in[28], *p2b0 = (const float*)d_in[29];
  const float* p2g0 = (const float*)d_in[30], *p2t0 = (const float*)d_in[31];
  const float* p2w1 = (const float*)d_in[32], *p2b1 = (const float*)d_in[33];
  const float* p2g1 = (const float*)d_in[34], *p2t1 = (const float*)d_in[35];
  const float* W3[5] = {(const float*)d_in[36], (const float*)d_in[40],
                        (const float*)d_in[44], (const float*)d_in[48],
                        (const float*)d_in[52]};
  const float* B3[5] = {(const float*)d_in[37], (const float*)d_in[41],
                        (const float*)d_in[45], (const float*)d_in[49],
                        (const float*)d_in[53]};
  const float* G3[4] = {(const float*)d_in[38], (const float*)d_in[42],
                        (const float*)d_in[46], (const float*)d_in[50]};
  const float* T3[4] = {(const float*)d_in[39], (const float*)d_in[43],
                        (const float*)d_in[47], (const float*)d_in[51]};

  // --- workspace layout (bump allocator, 256B aligned) ---
  char* base = (char*)d_ws;
  size_t off = 0;
  auto alloc = [&](size_t bytes) -> void* {
    void* p = base + off;
    off += (bytes + 255) & ~(size_t)255;
    return p;
  };
  float* dists = (float*)alloc((size_t)NPTS * 4);
  int*   idx1  = (int*)alloc((size_t)NC1 * 4);
  float* c1    = (float*)alloc((size_t)NC1 * 3 * 4);
  int*   grp1  = (int*)alloc((size_t)NC1 * NC1 * 4);
  float* f1    = (float*)alloc((size_t)NC1 * 128 * 4);
  int*   idx2  = (int*)alloc((size_t)NC2 * 4);
  float* c2    = (float*)alloc((size_t)NC2 * 3 * 4);
  int*   grp2  = (int*)alloc((size_t)NC2 * NC2 * 4);
  float* f2    = (float*)alloc((size_t)NC2 * 256 * 4);
  float* gx    = (float*)alloc((size_t)NC2 * 259 * 4);
  float* gh1   = (float*)alloc((size_t)NC2 * 256 * 4);
  float* gh2   = (float*)alloc((size_t)NC2 * 512 * 4);
  float* gh3   = (float*)alloc((size_t)NC2 * 1024 * 4);
  float* f3    = (float*)alloc((size_t)1024 * 4);
  float* x1    = (float*)alloc((size_t)NC2 * 1283 * 4);
  float* ybuf  = (float*)alloc((size_t)512 * 256 * 4);
  float* abuf  = (float*)alloc((size_t)512 * 256 * 4);
  float* acc   = (float*)alloc((size_t)512 * 4);
  float* f4    = (float*)alloc((size_t)NC2 * 256 * 4);
  int*   tki   = (int*)alloc((size_t)NC1 * 3 * 4);
  float* tkw   = (float*)alloc((size_t)NC1 * 3 * 4);
  float* x2    = (float*)alloc((size_t)NC1 * 387 * 4);
  float* f5    = (float*)alloc((size_t)NC1 * 128 * 4);
  int*   bki   = (int*)alloc((size_t)NPTS * 3 * 4);
  float* bkw   = (float*)alloc((size_t)NPTS * 3 * 4);
  __bf16* pB   = (__bf16*)alloc((size_t)160 * 128 * 2);
  __bf16* X0   = (__bf16*)alloc((size_t)NPTS * 160 * 2);
  __bf16* Xa   = (__bf16*)alloc((size_t)NPTS * 128 * 2);
  float* Ybig  = (float*)alloc((size_t)NPTS * 128 * 4);

  // ---------------- SA1: FPS -> ball query -> MLP(3,64,64,128) + maxpool
  fps_kernel<<<1, 1024, 0, stream>>>(coords, NPTS, NC1, idx1, c1, dists);
  ball_query<<<NC1, 32, 0, stream>>>(coords, NPTS, c1, 0.04f, NC1, grp1);
  sa_mlp<3, 64, 64, 128><<<NC1, 256, 0, stream>>>(
      coords, nullptr, c1, grp1, NC1, s1w0, s1b0, s1w1, s1b1, s1w2, s1b2, f1);

  // ---------------- SA2: FPS -> ball query -> MLP(131,128,128,256)
  fps_kernel<<<1, 1024, 0, stream>>>(c1, NC1, NC2, idx2, c2, dists);
  ball_query<<<NC2, 32, 0, stream>>>(c1, NC1, c2, 0.16f, NC2, grp2);
  sa_mlp<131, 128, 128, 256><<<NC2, 256, 0, stream>>>(
      c1, f1, c2, grp2, NC2, s2w0, s2b0, s2w1, s2b1, s2w2, s2b2, f2);

  // ---------------- Global MLP: [c2|f2](128,259) -> 256 -> 512 -> 1024 -> max
  fp_concat<<<(NC2 * 259 + 255) / 256, 256, 0, stream>>>(
      c2, f2, 256, nullptr, 0, nullptr, nullptr, gx, NC2, 259);
  dense_f32<<<(NC2 * 256 + 255) / 256, 256, 0, stream>>>(gx, 259, gw0, gb0, gh1, NC2, 256, 1);
  dense_f32<<<(NC2 * 512 + 255) / 256, 256, 0, stream>>>(gh1, 256, gw1, gb1, gh2, NC2, 512, 1);
  dense_f32<<<(NC2 * 1024 + 255) / 256, 256, 0, stream>>>(gh2, 512, gw2, gb2, gh3, NC2, 1024, 1);
  rowmax<<<4, 256, 0, stream>>>(gh3, NC2, 1024, f3);

  // ---------------- FP1: broadcast f3 -> [c2|f2|f3](128,1283) -> 256 BN ReLU -> 256 BN ReLU
  fp_concat<<<(NC2 * 1283 + 255) / 256, 256, 0, stream>>>(
      c2, f2, 256, f3, 1024, nullptr, nullptr, x1, NC2, 1283);
  dense_f32<<<(NC2 * 256 + 255) / 256, 256, 0, stream>>>(x1, 1283, p1w0, p1b0, ybuf, NC2, 256, 0);
  zero_f32<<<2, 256, 0, stream>>>(acc, 512);
  bn_stats<<<1, 256, 0, stream>>>(ybuf, NC2, 256, acc);
  bn_apply<true, false><<<(NC2 * 256 + 255) / 256, 256, 0, stream>>>(
      ybuf, acc, p1g0, p1t0, NC2, 256, abuf, nullptr);
  dense_f32<<<(NC2 * 256 + 255) / 256, 256, 0, stream>>>(abuf, 256, p1w1, p1b1, ybuf, NC2, 256, 0);
  zero_f32<<<2, 256, 0, stream>>>(acc, 512);
  bn_stats<<<1, 256, 0, stream>>>(ybuf, NC2, 256, acc);
  bn_apply<true, false><<<(NC2 * 256 + 255) / 256, 256, 0, stream>>>(
      ybuf, acc, p1g1, p1t1, NC2, 256, f4, nullptr);

  // ---------------- FP2: 3NN (c1 vs c2) -> [c1|f1|interp(f4)](512,387) -> 256 -> 128
  knn3<<<(NC1 + 255) / 256, 256, 0, stream>>>(c1, NC1, c2, NC2, tki, tkw);
  fp_concat<<<(NC1 * 387 + 255) / 256, 256, 0, stream>>>(
      c1, f1, 128, f4, 256, tki, tkw, x2, NC1, 387);
  dense_f32<<<(NC1 * 256 + 255) / 256, 256, 0, stream>>>(x2, 387, p2w0, p2b0, ybuf, NC1, 256, 0);
  zero_f32<<<2, 256, 0, stream>>>(acc, 512);
  bn_stats<<<(NC1 + 255) / 256, 256, 0, stream>>>(ybuf, NC1, 256, acc);
  bn_apply<true, false><<<(NC1 * 256 + 255) / 256, 256, 0, stream>>>(
      ybuf, acc, p2g0, p2t0, NC1, 256, abuf, nullptr);
  dense_f32<<<(NC1 * 128 + 255) / 256, 256, 0, stream>>>(abuf, 256, p2w1, p2b1, ybuf, NC1, 128, 0);
  zero_f32<<<2, 256, 0, stream>>>(acc, 512);
  bn_stats<<<(NC1 + 255) / 256, 256, 0, stream>>>(ybuf, NC1, 128, acc);
  bn_apply<true, false><<<(NC1 * 128 + 255) / 256, 256, 0, stream>>>(
      ybuf, acc, p2g1, p2t1, NC1, 128, f5, nullptr);

  // ---------------- FP3: 3NN (coords vs c1) -> WMMA MLP over N points
  knn3<<<NPTS / 256, 256, 0, stream>>>(coords, NPTS, c1, NC1, bki, bkw);
  fp3_concat_bf16<<<(NPTS * 160 + 255) / 256, 256, 0, stream>>>(coords, f5, bki, bkw, X0);

  const int Kreal[5] = {131, 128, 128, 128, 128};
  const int Kp[5]    = {160, 128, 128, 128, 128};
  const int Nreal[5] = {128, 128, 128, 128, 50};
  const int Np[5]    = {128, 128, 128, 128, 64};
  const int rowTiles = NPTS / 16;           // 8192
  const int gemmBlocks = rowTiles / 8;      // 8 waves per 256-thread block
  for (int l = 0; l < 5; ++l) {
    const long pbn = (long)Kp[l] * Np[l];
    pack_B<<<(int)((pbn + 255) / 256), 256, 0, stream>>>(W3[l], Kreal[l], Nreal[l],
                                                         Kp[l], Np[l], pB);
    const __bf16* Xin = (l == 0) ? X0 : Xa;
    float* Yout = (l == 4) ? (float*)d_out : Ybig;
    if (Kp[l] == 160) {
      wmma_dense<5><<<gemmBlocks, 256, 0, stream>>>(Xin, pB, B3[l], Yout,
                                                    NPTS, Np[l], Nreal[l]);
    } else {
      wmma_dense<4><<<gemmBlocks, 256, 0, stream>>>(Xin, pB, B3[l], Yout,
                                                    NPTS, Np[l], Nreal[l]);
    }
    if (l < 4) {
      zero_f32<<<1, 256, 0, stream>>>(acc, 256);
      bn_stats<<<NPTS / 256, 256, 0, stream>>>(Ybig, NPTS, 128, acc);
      bn_apply<true, true><<<(NPTS * 128) / 256, 256, 0, stream>>>(
          Ybig, acc, G3[l], T3[l], NPTS, 128, nullptr, Xa);
    }
  }
}
